// Attention_56178172232278
// MI455X (gfx1250) — matrix operand order repaired and checked by reading, not compiled, not run on hardware
//
#include <hip/hip_runtime.h>

// ---------- types ----------
typedef __bf16 bf16;
typedef __attribute__((ext_vector_type(16))) __bf16 v16bf;
typedef __attribute__((ext_vector_type(8)))  __bf16 v8bf;
typedef __attribute__((ext_vector_type(8)))  float  v8f;

#define B_   8
#define N_   1024
#define D_   512
#define H_   16
#define HD_  32
#define TBL_ 3969
#define M_   (B_ * N_)     // 8192 rows

// ---------- helpers ----------
__device__ __forceinline__ float redmax16(float v) {
#pragma unroll
  for (int m = 1; m < 16; m <<= 1) v = fmaxf(v, __shfl_xor(v, m, 32));
  return v;
}
__device__ __forceinline__ float redsum16(float v) {
#pragma unroll
  for (int m = 1; m < 16; m <<= 1) v += __shfl_xor(v, m, 32);
  return v;
}
__device__ __forceinline__ v16bf cat8(v8bf lo, v8bf hi) {
  return __builtin_shufflevector(lo, hi, 0,1,2,3,4,5,6,7,8,9,10,11,12,13,14,15);
}

// CDNA5 async copy: global -> LDS (16B per lane), tracked by ASYNCcnt.
// lds_off is the 32-bit LDS byte address (low 32 bits of generic LDS pointer).
__device__ __forceinline__ void async_b128(unsigned lds_off, const bf16* g) {
  asm volatile("global_load_async_to_lds_b128 %0, %1, off"
               :: "v"(lds_off), "v"(g) : "memory");
}

// ---------- f32 -> bf16 elementwise ----------
__global__ void k_f32_to_bf16(const float* __restrict__ in, bf16* __restrict__ out, int n) {
  int i = blockIdx.x * blockDim.x + threadIdx.x;
  if (i < n) out[i] = (bf16)in[i];
}

// ---------- CPB MLP: [TBL,2] -> relu(fc1) -> fc2 -> bias[H][TBL] ----------
__global__ void k_cpb(const float* __restrict__ ct, const float* __restrict__ w1,
                      const float* __restrict__ b1, const float* __restrict__ w2,
                      const float* __restrict__ b2, float* __restrict__ bias) {
  int t = blockIdx.x * blockDim.x + threadIdx.x;
  if (t >= TBL_ * H_) return;
  int h = t % H_, row = t / H_;
  float c0 = ct[row * 2 + 0], c1 = ct[row * 2 + 1];
  float acc = b2[h];
  for (int j = 0; j < 512; j++) {
    float hid = fmaxf(c0 * w1[j * 2 + 0] + c1 * w1[j * 2 + 1] + b1[j], 0.f);
    acc += hid * w2[h * 512 + j];
  }
  bias[h * TBL_ + row] = acc;
}

// ---------- WMMA GEMM with double-buffered async global->LDS copies ----------
// A[M,K] bf16 row-major  x  W[Nout,K] bf16 row-major (acts as B^T); out = A@W^T + bias.
// 64x64 block tile, 4 waves; per K-step each wave issues 4 async b128 copies for the
// NEXT tile, waits asynccnt<=4 (in-order completion => current tile landed), computes.
__global__ __launch_bounds__(128) void k_gemm(const bf16* __restrict__ A,
                                              const bf16* __restrict__ W,
                                              const float* __restrict__ bias,
                                              bf16* __restrict__ outb,
                                              float* __restrict__ outf,
                                              int M, int K, int Nout, int write_f32) {
  __shared__ __align__(64) bf16 sA[2][64 * 32];
  __shared__ __align__(64) bf16 sB[2][64 * 32];   // [o][k] = B column-major over k
  const int tid  = threadIdx.x;
  const int lane = tid & 31;
  const int wave = tid >> 5;
  const int hi   = (lane >> 4) & 1;
  const int l15  = lane & 15;
  const int m0 = blockIdx.x * 64;
  const int n0 = blockIdx.y * 64;

  v8f acc[4];
#pragma unroll
  for (int j = 0; j < 4; j++) acc[j] = {};

  const int ldrow = tid >> 1, ldseg = (tid & 1) * 16;   // 64 rows x 32 bf16, 32B/thread
  const bf16* gA = A + (size_t)(m0 + ldrow) * K + ldseg;
  const bf16* gB = W + (size_t)(n0 + ldrow) * K + ldseg;
  const unsigned la[2] = { (unsigned)(uintptr_t)&sA[0][ldrow * 32 + ldseg],
                           (unsigned)(uintptr_t)&sA[1][ldrow * 32 + ldseg] };
  const unsigned lb[2] = { (unsigned)(uintptr_t)&sB[0][ldrow * 32 + ldseg],
                           (unsigned)(uintptr_t)&sB[1][ldrow * 32 + ldseg] };

  // prologue: stage K-step 0 into buffer 0
  async_b128(la[0],      gA);
  async_b128(la[0] + 16, gA + 8);
  async_b128(lb[0],      gB);
  async_b128(lb[0] + 16, gB + 8);

  const int nk = K >> 5;
  for (int kk = 0; kk < nk; ++kk) {
    const int cur = kk & 1;
    if (kk + 1 < nk) {
      const bf16* nA = gA + (size_t)(kk + 1) * 32;
      const bf16* nB = gB + (size_t)(kk + 1) * 32;
      async_b128(la[cur ^ 1],      nA);
      async_b128(la[cur ^ 1] + 16, nA + 8);
      async_b128(lb[cur ^ 1],      nB);
      async_b128(lb[cur ^ 1] + 16, nB + 8);
      asm volatile("s_wait_asynccnt 0x4" ::: "memory");  // current tile complete
    } else {
      asm volatile("s_wait_asynccnt 0x0" ::: "memory");
    }
    __syncthreads();

    const bf16* cA = sA[cur];
    const bf16* cB = sB[cur];
    // A fragment (16x32): lane row = lane&15; K = e + 16*hi
    const int arow = wave * 16 + l15;
    v8bf alo = *(const v8bf*)&cA[arow * 32 + 16 * hi];
    v8bf ahi = *(const v8bf*)&cA[arow * 32 + 16 * hi + 8];
    v16bf af = cat8(alo, ahi);
    // B fragments (32x16): col = lane&15, K = e + 16*hi -> contiguous in sB[o][k]
    v16bf bf0 = *(const v16bf*)&cB[(0 * 16 + l15) * 32 + 16 * hi];
    v16bf bf1 = *(const v16bf*)&cB[(1 * 16 + l15) * 32 + 16 * hi];
    v16bf bf2 = *(const v16bf*)&cB[(2 * 16 + l15) * 32 + 16 * hi];
    v16bf bf3 = *(const v16bf*)&cB[(3 * 16 + l15) * 32 + 16 * hi];
    acc[0] = __builtin_amdgcn_wmma_f32_16x16x32_bf16(false, af, false, bf0, (short)0, acc[0], false, false);
    acc[1] = __builtin_amdgcn_wmma_f32_16x16x32_bf16(false, af, false, bf1, (short)0, acc[1], false, false);
    acc[2] = __builtin_amdgcn_wmma_f32_16x16x32_bf16(false, af, false, bf2, (short)0, acc[2], false, false);
    acc[3] = __builtin_amdgcn_wmma_f32_16x16x32_bf16(false, af, false, bf3, (short)0, acc[3], false, false);
    __syncthreads();
  }

#pragma unroll
  for (int j = 0; j < 4; j++) {
    const int col = n0 + j * 16 + l15;
    const float bs = bias ? bias[col] : 0.f;
#pragma unroll
    for (int r = 0; r < 8; r++) {
      const int row = m0 + wave * 16 + r + 8 * hi;   // C layout: M = r + 8*hi
      const float v = acc[j][r] + bs;
      if (write_f32) outf[(size_t)row * Nout + col] = v;
      else           outb[(size_t)row * Nout + col] = (bf16)v;
    }
  }
}

// ---------- qkv post: l2norm, query-embedding, softplus(temp)*log(N) scale; v transpose ----------
__global__ void k_qkv_post(const bf16* __restrict__ qkv, const float* __restrict__ temp,
                           const float* __restrict__ qe, const float* __restrict__ sls,
                           bf16* __restrict__ qn, bf16* __restrict__ kn, bf16* __restrict__ vT) {
  int t = blockIdx.x * blockDim.x + threadIdx.x;
  if (t >= B_ * H_ * N_) return;
  int n = t % N_, h = (t / N_) % H_, b = t / (N_ * H_);
  const size_t base = (size_t)(b * N_ + n) * (3 * D_) + h * HD_;
  float q[HD_], k[HD_], sq = 0.f, sk = 0.f;
#pragma unroll
  for (int d = 0; d < HD_; d++) { q[d] = (float)qkv[base + d];       sq += q[d] * q[d]; }
#pragma unroll
  for (int d = 0; d < HD_; d++) { k[d] = (float)qkv[base + D_ + d];  sk += k[d] * k[d]; }
  const float qinv = 1.f / fmaxf(sqrtf(sq), 1e-12f);
  const float kinv = 1.f / fmaxf(sqrtf(sk), 1e-12f);
  const float scale = logf(1.f + __expf(temp[h])) * sls[0];   // softplus * seq_length_scale
  const size_t ob = ((size_t)(b * H_ + h) * N_ + n) * HD_;
#pragma unroll
  for (int d = 0; d < HD_; d++) qn[ob + d] = (bf16)((q[d] * qinv + qe[h * HD_ + d]) * scale);
#pragma unroll
  for (int d = 0; d < HD_; d++) kn[ob + d] = (bf16)(k[d] * kinv);
  const size_t vb = (size_t)(b * H_ + h) * (HD_ * N_);
#pragma unroll
  for (int d = 0; d < HD_; d++) vT[vb + (size_t)d * N_ + n] = qkv[base + 2 * D_ + d];
}

// ---------- flash attention: one wave per (b, h, 16-query tile); key blocks of 32 ----------
__global__ __launch_bounds__(256) void k_attn(const bf16* __restrict__ qn,
                                              const bf16* __restrict__ kn,
                                              const bf16* __restrict__ vT,
                                              const float* __restrict__ bias,
                                              const int* __restrict__ rpi,
                                              bf16* __restrict__ out) {
  __shared__ __align__(64) bf16 pbuf[8][16 * 32];
  const int lane = threadIdx.x & 31;
  const int wave = threadIdx.x >> 5;
  const int hi   = (lane >> 4) & 1;
  const int l15  = lane & 15;
  const int tile = blockIdx.x * 8 + wave;        // 8192 tiles total
  const int qt = tile & 63, h = (tile >> 6) & 15, b = tile >> 10;
  const int qbase = qt * 16;
  const size_t bh = (size_t)(b * H_ + h);
  const bf16* qp = qn + bh * N_ * HD_;
  const bf16* kp = kn + bh * N_ * HD_;
  const bf16* vp = vT + bh * HD_ * N_;
  const float* bt = bias + (size_t)h * TBL_;

  // Q as A fragment (rows = queries, K = hd 0..31)
  v8bf qlo = *(const v8bf*)&qp[(qbase + l15) * HD_ + 16 * hi];
  v8bf qhi = *(const v8bf*)&qp[(qbase + l15) * HD_ + 16 * hi + 8];
  v16bf qf = cat8(qlo, qhi);

  v8f o0 = {}, o1 = {};
  float mrow[8], lrow[8];
#pragma unroll
  for (int r = 0; r < 8; r++) { mrow[r] = -3.0e38f; lrow[r] = 0.f; }

  bf16* pb = pbuf[wave];

  for (int j = 0; j < N_ / 32; j++) {
    const int kb = j * 32;
    if (j + 1 < N_ / 32)
      __builtin_prefetch(&kp[(kb + 32 + l15) * HD_], 0, 0);

    // K as B fragments (K = hd, cols = 16 keys each)
    v16bf kf0 = *(const v16bf*)&kp[(kb + l15) * HD_ + 16 * hi];
    v16bf kf1 = *(const v16bf*)&kp[(kb + 16 + l15) * HD_ + 16 * hi];
    v8f z0 = {}, z1 = {};
    v8f s0 = __builtin_amdgcn_wmma_f32_16x16x32_bf16(false, qf, false, kf0, (short)0, z0, false, false);
    v8f s1 = __builtin_amdgcn_wmma_f32_16x16x32_bf16(false, qf, false, kf1, (short)0, z1, false, false);

    // relative position bias: rb[h,q,k] = bias[h, rpi[q*N + k]]  (identity resizes cancel)
#pragma unroll
    for (int r = 0; r < 8; r++) {
      const int qrow = qbase + r + 8 * hi;
      s0[r] += bt[rpi[(size_t)qrow * N_ + kb + l15]];
      s1[r] += bt[rpi[(size_t)qrow * N_ + kb + 16 + l15]];
    }

    // online softmax (row lives across the 16-lane half-group)
#pragma unroll
    for (int r = 0; r < 8; r++) {
      const float rm = redmax16(fmaxf(s0[r], s1[r]));
      const float nm = fmaxf(mrow[r], rm);
      const float alpha = __expf(mrow[r] - nm);
      const float p0 = __expf(s0[r] - nm);
      const float p1 = __expf(s1[r] - nm);
      lrow[r] = lrow[r] * alpha + redsum16(p0 + p1);
      mrow[r] = nm;
      o0[r] *= alpha;
      o1[r] *= alpha;
      pb[(r + 8 * hi) * 32 + l15]      = (bf16)p0;   // stage P row-major [16 q][32 keys]
      pb[(r + 8 * hi) * 32 + 16 + l15] = (bf16)p1;
    }
    asm volatile("s_wait_dscnt 0x0" ::: "memory");   // LDS in-order within wave; wait staged P

    // P as A fragment (rows = queries, K = 32 keys)
    v8bf plo = *(const v8bf*)&pb[l15 * 32 + 16 * hi];
    v8bf phi = *(const v8bf*)&pb[l15 * 32 + 16 * hi + 8];
    v16bf pf = cat8(plo, phi);

    // V as B fragments from vT[hd][n]: element e -> key kb + e + 16*hi (contiguous)
    v16bf vf0 = *(const v16bf*)&vp[(size_t)l15 * N_ + kb + 16 * hi];
    v16bf vf1 = *(const v16bf*)&vp[(size_t)(16 + l15) * N_ + kb + 16 * hi];
    o0 = __builtin_amdgcn_wmma_f32_16x16x32_bf16(false, pf, false, vf0, (short)0, o0, false, false);
    o1 = __builtin_amdgcn_wmma_f32_16x16x32_bf16(false, pf, false, vf1, (short)0, o1, false, false);
  }

#pragma unroll
  for (int r = 0; r < 8; r++) {
    const float inv = 1.f / lrow[r];
    const int qrow = qbase + r + 8 * hi;
    const size_t ob = ((size_t)(b * N_) + qrow) * D_ + h * HD_;
    out[ob + l15]      = (bf16)(o0[r] * inv);
    out[ob + 16 + l15] = (bf16)(o1[r] * inv);
  }
}

// ---------- launch ----------
extern "C" void kernel_launch(void* const* d_in, const int* in_sizes, int n_in,
                              void* d_out, int out_size, void* d_ws, size_t ws_size,
                              hipStream_t stream) {
  (void)in_sizes; (void)n_in; (void)out_size; (void)ws_size;
  const float* x      = (const float*)d_in[0];
  const float* qkv_w  = (const float*)d_in[1];
  const float* qkv_b  = (const float*)d_in[2];
  const float* proj_w = (const float*)d_in[3];
  const float* proj_b = (const float*)d_in[4];
  const float* temp   = (const float*)d_in[5];
  const float* qe     = (const float*)d_in[6];
  const float* c1w    = (const float*)d_in[7];
  const float* c1b    = (const float*)d_in[8];
  const float* c2w    = (const float*)d_in[9];
  const float* c2b    = (const float*)d_in[10];
  const float* rct    = (const float*)d_in[11];
  const int*   rpi    = (const int*)d_in[12];
  const float* sls    = (const float*)d_in[13];
  float* out = (float*)d_out;

  auto al = [](size_t s) { return (s + 255) & ~(size_t)255; };
  char* w = (char*)d_ws;
  bf16* xbf    = (bf16*)w;  w += al((size_t)M_ * D_ * 2);
  bf16* wqkvbf = (bf16*)w;  w += al((size_t)3 * D_ * D_ * 2);
  bf16* wprjbf = (bf16*)w;  w += al((size_t)D_ * D_ * 2);
  bf16* qkvbf  = (bf16*)w;  w += al((size_t)M_ * 3 * D_ * 2);
  bf16* qnb    = (bf16*)w;  w += al((size_t)B_ * H_ * N_ * HD_ * 2);
  bf16* knb    = (bf16*)w;  w += al((size_t)B_ * H_ * N_ * HD_ * 2);
  bf16* vTb    = (bf16*)w;  w += al((size_t)B_ * H_ * HD_ * N_ * 2);
  bf16* aob    = (bf16*)w;  w += al((size_t)M_ * D_ * 2);
  float* biasT = (float*)w; w += al((size_t)H_ * TBL_ * 4);

  k_f32_to_bf16<<<(M_ * D_ + 255) / 256, 256, 0, stream>>>(x, xbf, M_ * D_);
  k_f32_to_bf16<<<(3 * D_ * D_ + 255) / 256, 256, 0, stream>>>(qkv_w, wqkvbf, 3 * D_ * D_);
  k_f32_to_bf16<<<(D_ * D_ + 255) / 256, 256, 0, stream>>>(proj_w, wprjbf, D_ * D_);
  k_cpb<<<(TBL_ * H_ + 127) / 128, 128, 0, stream>>>(rct, c1w, c1b, c2w, c2b, biasT);

  dim3 g1(M_ / 64, (3 * D_) / 64);
  k_gemm<<<g1, 128, 0, stream>>>(xbf, wqkvbf, qkv_b, qkvbf, nullptr, M_, D_, 3 * D_, 0);

  k_qkv_post<<<(B_ * H_ * N_ + 255) / 256, 256, 0, stream>>>(qkvbf, temp, qe, sls, qnb, knb, vTb);

  k_attn<<<(B_ * H_ * (N_ / 16)) / 8, 256, 0, stream>>>(qnb, knb, vTb, biasT, rpi, aob);

  dim3 g2(M_ / 64, D_ / 64);
  k_gemm<<<g2, 128, 0, stream>>>(aob, wprjbf, proj_b, nullptr, out, M_, D_, D_, 1);
}
